// MultiHeadAttention_3496103379267
// MI455X (gfx1250) — matrix operand order repaired and checked by reading, not compiled, not run on hardware
//
#include <hip/hip_runtime.h>
#include <hip/hip_bf16.h>

// ---------- problem constants ----------
#define DMODEL 1024
#define NHEADS 16
#define DK     64
#define SEQ    2048
#define BATCH  2
#define MTOT   (BATCH * SEQ)   // 4096

// Q is pre-scaled by 1/sqrt(dk) * log2(e) so softmax runs in base-2 domain.
#define QSCALE 0.18033688011112042f   // 0.125 * 1.4426950408889634

// ---------- CDNA5 WMMA types ----------
typedef __attribute__((ext_vector_type(16))) __bf16 v16bf;
typedef __attribute__((ext_vector_type(8)))  __bf16 v8bf;
typedef __attribute__((ext_vector_type(8)))  float  v8f;

static __device__ __forceinline__ unsigned short f2bf_rne(float f) {
    unsigned u = __builtin_bit_cast(unsigned, f);
    unsigned r = u + 0x7FFFu + ((u >> 16) & 1u);
    return (unsigned short)(r >> 16);
}
// round-half-up: 2 VALU ops, differs from RNE only on exact ties
static __device__ __forceinline__ unsigned short f2bf_fast(float f) {
    unsigned u = __builtin_bit_cast(unsigned, f);
    return (unsigned short)((u + 0x8000u) >> 16);
}

// 16-lane XOR butterfly via ds_swizzle (group-of-32: and=0x1F, xor=M).
// Masks 1/2/4/8 never cross the 16-lane half boundary.
template<int M>
static __device__ __forceinline__ float swz_xor(float v) {
    int i = __builtin_bit_cast(int, v);
    i = __builtin_amdgcn_ds_swizzle(i, 0x1F | (M << 10));
    return __builtin_bit_cast(float, i);
}
static __device__ __forceinline__ float groupmax16(float v) {
    v = fmaxf(v, swz_xor<1>(v));
    v = fmaxf(v, swz_xor<2>(v));
    v = fmaxf(v, swz_xor<4>(v));
    v = fmaxf(v, swz_xor<8>(v));
    return v;
}
static __device__ __forceinline__ float groupsum16(float v) {
    v += swz_xor<1>(v);
    v += swz_xor<2>(v);
    v += swz_xor<4>(v);
    v += swz_xor<8>(v);
    return v;
}

// A-matrix fragment (16x32 bf16), contiguous k order:
// lanes 0-15: row=lane,   elems 0..15 = K[k0..k0+15]
// lanes16-31: row=lane-16,elems 0..15 = K[k0+16..k0+31]
static __device__ __forceinline__ v16bf load_a_frag(const unsigned short* p, int half) {
    v8bf lo = *(const v8bf*)(p + half * 16);
    v8bf hi = *(const v8bf*)(p + half * 16 + 8);
    return __builtin_shufflevector(lo, hi, 0,1,2,3,4,5,6,7,8,9,10,11,12,13,14,15);
}

static __device__ __forceinline__ v8f wmma_bf16(v16bf a, v16bf b, v8f c) {
    return __builtin_amdgcn_wmma_f32_16x16x32_bf16(
        false, a, false, b, (short)0, c, false, false);
}

// ---------- stage 1: fp32 -> bf16 ----------
__global__ void cvt_bf16_kernel(const float* __restrict__ src,
                                unsigned short* __restrict__ dst, int n) {
    int i = (blockIdx.x * blockDim.x + threadIdx.x) * 4;
    if (i + 3 < n) {
        float4 v = *(const float4*)(src + i);
        dst[i + 0] = f2bf_rne(v.x);
        dst[i + 1] = f2bf_rne(v.y);
        dst[i + 2] = f2bf_rne(v.z);
        dst[i + 3] = f2bf_rne(v.w);
    }
}

// ---------- stage 2/4: C[m,n] = oscale * sum_k A[m,k] * W[n,k]
// MODE 0: bf16 out, head layout [B,H,S,dk]      (Q with QSCALE, K)
// MODE 1: bf16 out, transposed  [B,H,dk,S]      (V)
// MODE 2: fp32 out, plain [M,N]                 (final projection)
// Block: 256 thr = 8 waves, block tile 128x128, wave tile 32Mx64N.
template<int MODE>
__global__ __launch_bounds__(256)
void gemm_bf16_kernel(const unsigned short* __restrict__ A,
                      const unsigned short* __restrict__ W,
                      float* __restrict__ outf,
                      unsigned short* __restrict__ outb,
                      int M, int N, int K, float oscale) {
    const int lane = threadIdx.x & 31;
    const int wave = threadIdx.x >> 5;
    const int half = lane >> 4;
    const int l15  = lane & 15;
    const int wm = wave & 3;          // 4 waves along M
    const int wn = wave >> 2;         // 2 waves along N
    const int mbase = blockIdx.y * 128 + wm * 32;
    const int nbase = blockIdx.x * 128 + wn * 64;

    v8f c[2][4];
#pragma unroll
    for (int i = 0; i < 2; ++i)
#pragma unroll
        for (int j = 0; j < 4; ++j) c[i][j] = (v8f){0.f,0.f,0.f,0.f,0.f,0.f,0.f,0.f};

    const unsigned short* arow0 = A + (size_t)(mbase + l15) * K;
    const unsigned short* arow1 = A + (size_t)(mbase + 16 + l15) * K;
    const unsigned short* wrow[4];
#pragma unroll
    for (int j = 0; j < 4; ++j)
        wrow[j] = W + (size_t)(nbase + j * 16 + l15) * K;

    for (int k0 = 0; k0 < K; k0 += 32) {
        if (k0 + 128 < K) {                         // gfx1250 global_prefetch_b8
            __builtin_prefetch(arow0 + k0 + 128, 0, 1);
            __builtin_prefetch(wrow[0] + k0 + 128, 0, 1);
        }
        v16bf a0 = load_a_frag(arow0 + k0, half);
        v16bf a1 = load_a_frag(arow1 + k0, half);
        v16bf b[4];
#pragma unroll
        for (int j = 0; j < 4; ++j)
            b[j] = *(const v16bf*)(wrow[j] + k0 + half * 16);   // 32B contiguous
#pragma unroll
        for (int j = 0; j < 4; ++j) {
            c[0][j] = wmma_bf16(a0, b[j], c[0][j]);
            c[1][j] = wmma_bf16(a1, b[j], c[1][j]);
        }
    }

#pragma unroll
    for (int i = 0; i < 2; ++i)
#pragma unroll
        for (int j = 0; j < 4; ++j)
#pragma unroll
            for (int r = 0; r < 8; ++r) {
                int m = mbase + i * 16 + r + half * 8;
                int n = nbase + j * 16 + l15;
                float val = c[i][j][r] * oscale;
                if (MODE == 2) {
                    outf[(size_t)m * N + n] = val;
                } else {
                    int bb = m >> 11, s = m & (SEQ - 1);
                    int h  = n >> 6,  d = n & (DK - 1);
                    size_t idx = (MODE == 0)
                        ? ((size_t)((bb * NHEADS + h) * SEQ + s)) * DK + d
                        : ((size_t)((bb * NHEADS + h) * DK + d)) * SEQ + s;
                    outb[idx] = f2bf_rne(val);
                }
            }
    (void)outf; (void)outb;
}

// ---------- stage 3: fused attention (flash-style, one 16-row Q tile / wave) ----------
// Q (pre-scaled), K: [B,H,S,dk] bf16   Vt: [B,H,dk,S] bf16   Ob: [B,S,D] bf16
// Key tile = 64 per iteration: 8 score WMMAs + 8 PV WMMAs.
__global__ __launch_bounds__(128)
void attn_kernel(const unsigned short* __restrict__ Q,
                 const unsigned short* __restrict__ Km,
                 const unsigned short* __restrict__ Vt,
                 unsigned short* __restrict__ Ob) {
    __shared__ unsigned short plds[4][16 * 64];    // per-wave P transpose patch

    const int lane = threadIdx.x & 31;
    const int wave = threadIdx.x >> 5;
    const int half = lane >> 4;
    const int l15  = lane & 15;
    const int gw = blockIdx.x * 4 + wave;          // 0..4095
    const int qt = gw & 127;                       // q tile within (b,h)
    const int bh = gw >> 7;                        // b*16 + h

    // Q A-fragments (16x64 -> two 16x32 frags), resident for the whole loop
    const size_t qoff = ((size_t)bh * SEQ + qt * 16 + l15) * DK;
    const v16bf aq0 = load_a_frag(Q + qoff, half);
    const v16bf aq1 = load_a_frag(Q + qoff + 32, half);

    v8f o[4];
#pragma unroll
    for (int f = 0; f < 4; ++f) o[f] = (v8f){0.f,0.f,0.f,0.f,0.f,0.f,0.f,0.f};
    float mrow[8], lsum[8];                        // lsum is lane-local (reduced at end)
#pragma unroll
    for (int r = 0; r < 8; ++r) { mrow[r] = -1e30f; lsum[r] = 0.f; }

    const unsigned short* kbase = Km + (size_t)bh * SEQ * DK;
    const unsigned short* vbase = Vt + (size_t)bh * DK * SEQ;
    unsigned short* myp = &plds[wave][0];

    for (int kt = 0; kt < SEQ; kt += 64) {
        // ---- scores: S[16 x 64] = Qs(16x64) @ K^T(64x64), four 16-col C frags
        v8f sc[4];
#pragma unroll
        for (int kk = 0; kk < 4; ++kk) {
            const unsigned short* kr = kbase + (size_t)(kt + kk * 16 + l15) * DK;
            v16bf b0 = *(const v16bf*)(kr + half * 16);        // d = 0..31
            v16bf b1 = *(const v16bf*)(kr + 32 + half * 16);   // d = 32..63
            v8f z = (v8f){0.f,0.f,0.f,0.f,0.f,0.f,0.f,0.f};
            z = wmma_bf16(aq0, b0, z);
            z = wmma_bf16(aq1, b1, z);
            sc[kk] = z;
        }
        // ---- online softmax (base-2 domain; row = r + half*8)
#pragma unroll
        for (int r = 0; r < 8; ++r) {
            float s0 = sc[0][r], s1 = sc[1][r], s2 = sc[2][r], s3 = sc[3][r];
            float v = groupmax16(fmaxf(fmaxf(s0, s1), fmaxf(s2, s3)));
            float mnew  = fmaxf(mrow[r], v);
            float alpha = exp2f(mrow[r] - mnew);
            mrow[r] = mnew;
            float p0 = exp2f(s0 - mnew);
            float p1 = exp2f(s1 - mnew);
            float p2 = exp2f(s2 - mnew);
            float p3 = exp2f(s3 - mnew);
            lsum[r] = lsum[r] * alpha + ((p0 + p1) + (p2 + p3));
            o[0][r] *= alpha; o[1][r] *= alpha; o[2][r] *= alpha; o[3][r] *= alpha;
            // stage P (bf16) into wave-private LDS for the layout transpose
            unsigned short* pr = myp + (r + half * 8) * 64 + l15;
            pr[0]  = f2bf_fast(p0);
            pr[16] = f2bf_fast(p1);
            pr[32] = f2bf_fast(p2);
            pr[48] = f2bf_fast(p3);
        }
        // ---- reload P as two A fragments (16x32 each, K = key index)
        const unsigned short* prow = myp + l15 * 64;
        v16bf ap0 = load_a_frag(prow, half);
        v16bf ap1 = load_a_frag(prow + 32, half);
        // ---- O += P @ V  (V transposed: keys contiguous per d-row)
#pragma unroll
        for (int f = 0; f < 4; ++f) {
            const unsigned short* vr = vbase + (size_t)(f * 16 + l15) * SEQ + kt + half * 16;
            v16bf bv0 = *(const v16bf*)vr;
            v16bf bv1 = *(const v16bf*)(vr + 32);
            o[f] = wmma_bf16(ap0, bv0, o[f]);
            o[f] = wmma_bf16(ap1, bv1, o[f]);
        }
    }

    // ---- finalize: reduce lane-local sums, O /= l, store to [B,S,D] bf16
    const int b = bh >> 4, h = bh & 15;
#pragma unroll
    for (int r = 0; r < 8; ++r) {
        float inv = 1.0f / groupsum16(lsum[r]);
        int srow = qt * 16 + r + half * 8;
        size_t base = ((size_t)b * SEQ + srow) * DMODEL + h * DK;
#pragma unroll
        for (int f = 0; f < 4; ++f)
            Ob[base + f * 16 + l15] = f2bf_rne(o[f][r] * inv);
    }
}

// ---------- host-side launch ----------
extern "C" void kernel_launch(void* const* d_in, const int* in_sizes, int n_in,
                              void* d_out, int out_size, void* d_ws, size_t ws_size,
                              hipStream_t stream) {
    const float* x  = (const float*)d_in[0];
    const float* Wq = (const float*)d_in[1];
    const float* Wk = (const float*)d_in[2];
    const float* Wv = (const float*)d_in[3];
    const float* Wo = (const float*)d_in[4];

    unsigned short* ws = (unsigned short*)d_ws;
    // offsets in ushort elements
    unsigned short* xb  = ws;                         // 4096*1024
    unsigned short* wqb = ws + 4194304;               // 1024*1024 each
    unsigned short* wkb = ws + 5242880;
    unsigned short* wvb = ws + 6291456;
    unsigned short* wob = ws + 7340032;
    unsigned short* qb  = ws + 8388608;               // [B,H,S,dk] (pre-scaled)
    unsigned short* kb  = ws + 12582912;              // [B,H,S,dk]
    unsigned short* vtb = ws + 16777216;              // [B,H,dk,S]
    unsigned short* aob = ws + 20971520;              // [B,S,D]
    (void)in_sizes; (void)n_in; (void)out_size; (void)ws_size;

    // stage 1: convert to bf16
    cvt_bf16_kernel<<<(MTOT * DMODEL) / 1024, 256, 0, stream>>>(x,  xb,  MTOT * DMODEL);
    cvt_bf16_kernel<<<(DMODEL * DMODEL) / 1024, 256, 0, stream>>>(Wq, wqb, DMODEL * DMODEL);
    cvt_bf16_kernel<<<(DMODEL * DMODEL) / 1024, 256, 0, stream>>>(Wk, wkb, DMODEL * DMODEL);
    cvt_bf16_kernel<<<(DMODEL * DMODEL) / 1024, 256, 0, stream>>>(Wv, wvb, DMODEL * DMODEL);
    cvt_bf16_kernel<<<(DMODEL * DMODEL) / 1024, 256, 0, stream>>>(Wo, wob, DMODEL * DMODEL);

    // stage 2: projections (M=4096, N=1024, K=1024; block tile 128x128)
    dim3 ggrid(DMODEL / 128, MTOT / 128);
    gemm_bf16_kernel<0><<<ggrid, 256, 0, stream>>>(xb, wqb, nullptr, qb,  MTOT, DMODEL, DMODEL, QSCALE);
    gemm_bf16_kernel<0><<<ggrid, 256, 0, stream>>>(xb, wkb, nullptr, kb,  MTOT, DMODEL, DMODEL, 1.0f);
    gemm_bf16_kernel<1><<<ggrid, 256, 0, stream>>>(xb, wvb, nullptr, vtb, MTOT, DMODEL, DMODEL, 1.0f);

    // stage 3: fused attention. 4096 waves (one 16-row Q tile each), 4 waves/block.
    attn_kernel<<<(BATCH * NHEADS * (SEQ / 16)) / 4, 128, 0, stream>>>(qb, kb, vtb, aob);

    // stage 4: output projection -> fp32 d_out
    gemm_bf16_kernel<2><<<ggrid, 256, 0, stream>>>(aob, wob, (float*)d_out, nullptr,
                                                   MTOT, DMODEL, DMODEL, 1.0f);
}
